// Model_21157008900232
// MI455X (gfx1250) — matrix operand order repaired and checked by reading, not compiled, not run on hardware
//
#include <hip/hip_runtime.h>
#include <hip/hip_bf16.h>
#include <stdint.h>

// MI455X / gfx1250, wave32. Conv1d as implicit GEMM on v_wmma_f32_16x16x32_bf16,
// with weight tiles streamed by the Tensor Data Mover (tensor_load_to_lds),
// double-buffered against TENSORcnt.
typedef __bf16 bf16;
typedef __attribute__((ext_vector_type(16))) __bf16 v16bf;
typedef __attribute__((ext_vector_type(8)))  __bf16 v8bf;
typedef __attribute__((ext_vector_type(8)))  float  v8f;
typedef __attribute__((ext_vector_type(4)))  unsigned u32x4;
typedef __attribute__((ext_vector_type(8)))  unsigned u32x8;

#define SIGLEN 400
#define BEATL  80
#define STRIDE0 40
#define TBEATS 9
#define LEADS  12

// ---------------------------------------------------------------------------
// Weight repack: (Cout, CinK) f32 -> (Cout, Kpad) bf16, zero padded.
// ---------------------------------------------------------------------------
__global__ void pack_w_kernel(const float* __restrict__ w, bf16* __restrict__ dst,
                              int Cout, int CinK, int Kpad) {
    int i = blockIdx.x * 256 + threadIdx.x;
    int tot = Cout * Kpad;
    if (i >= tot) return;
    int r = i / Kpad, k = i - r * Kpad;
    float v = (k < CinK) ? w[(long)r * CinK + k] : 0.0f;
    dst[i] = (bf16)v;
}

// ---------------------------------------------------------------------------
// Beat extraction: x (nNodes, 400) f32 -> seg (nNodes*9, 80) bf16
// ---------------------------------------------------------------------------
__global__ void extract_beats_kernel(const float* __restrict__ x, bf16* __restrict__ seg,
                                     int nNodes) {
    int i = blockIdx.x * 256 + threadIdx.x;
    int tot = nNodes * TBEATS * BEATL;
    if (i >= tot) return;
    int j = i % BEATL;
    int t = (i / BEATL) % TBEATS;
    int n = i / (BEATL * TBEATS);
    seg[i] = (bf16)x[(long)n * SIGLEN + t * STRIDE0 + j];
}

// ---------------------------------------------------------------------------
// Implicit-GEMM conv1d with WMMA + TDM weight streaming.
//   out[n][co][t] = sum_{ci,dk} w[co][ci*k+dk] * in[n][ci][t*stride+dk-pad]
// Block: 256 threads (8 wave32), tile = Cout x 64 output columns.
// Weights: TDM tensor_load_to_lds, 2-D tile (32 x Cout, row stride Kpad),
//          double-buffered (issue ks+1, wait TENSORcnt<=1 for ks).
// Im2col:  VALU gather (zero-padded) into LDS.
// Epilogue modes: 0=relu, 1=relu+bn, 2=add-ident+relu+bn, 3=raw
// ---------------------------------------------------------------------------
__launch_bounds__(256)
__global__ void conv_wmma_kernel(
    const bf16* __restrict__ in, const bf16* __restrict__ wts,
    const bf16* __restrict__ ident,
    const float* __restrict__ bng, const float* __restrict__ bnb,
    const float* __restrict__ bnm, const float* __restrict__ bnv,
    bf16* __restrict__ out,
    int nBeats, int Cin, int Cout, int Lin, int Lout,
    int ktaps, int stride, int pad, int Ktrue, int Kpad, int mode)
{
    __shared__ __align__(64) bf16 As[2][128 * 32]; // double-buffered weight chunk
    __shared__ __align__(64) bf16 Bs[64 * 32];     // im2col chunk [col][k]

    const int tid  = threadIdx.x;
    const int lane = tid & 31;
    const int w    = tid >> 5;        // wave id 0..7
    const int h    = lane >> 4;       // lane half
    const int colBase = blockIdx.x * 64;

    // im2col gather mapping (8 contiguous K values of one column per thread)
    const int cLocal = tid >> 2;              // 0..63
    const int kSub   = (tid & 3) * 8;         // 0,8,16,24
    const int gcol   = colBase + cLocal;
    const int nIdx   = gcol / Lout;
    const int tPos   = gcol - nIdx * Lout;

    const int mTiles = Cout >> 4;
    const int kSteps = Kpad >> 5;

    // ---- TDM descriptor group1 (loop invariant) ----
    // dw0: workgroup_mask=0, data_size=1 (2 bytes)
    // dw1[31:16]=tensor_dim0[15:0]; dw2[15:0]=tensor_dim0[31:16]
    // dw2[31:16]=tensor_dim1[15:0]; dw3[31:16]=tile_dim0=32
    // dw4[15:0]=tile_dim1=Cout;     dw5=tensor_dim0_stride=Kpad
    u32x8 g1;
    g1[0] = 0x10000u;
    g1[1] = ((unsigned)Kpad & 0xffffu) << 16;
    g1[2] = (((unsigned)Kpad >> 16) & 0xffffu) | (((unsigned)Cout & 0xffffu) << 16);
    g1[3] = (32u << 16);
    g1[4] = (unsigned)Cout & 0xffffu;
    g1[5] = (unsigned)Kpad;
    g1[6] = 0u;
    g1[7] = 0u;
    // LDS byte addresses (low 32 bits of generic pointer == LDS offset)
    const unsigned ldsA0 = (unsigned)(size_t)&As[0][0];
    const unsigned ldsA1 = (unsigned)(size_t)&As[1][0];

    v8f zero8 = {0.f,0.f,0.f,0.f,0.f,0.f,0.f,0.f};
    v8f acc[4];
    acc[0] = zero8; acc[1] = zero8; acc[2] = zero8; acc[3] = zero8;

    // issue first weight chunk (wave 0 only; TDM ignores EXEC, wave-uniform branch)
    if (w == 0) {
        unsigned long long ga = (unsigned long long)(size_t)wts;
        u32x4 g0;
        g0[0] = 1u;                                   // count=1 valid descriptor
        g0[1] = ldsA0;                                // lds_addr
        g0[2] = (unsigned)ga;                         // global_addr[31:0]
        g0[3] = (unsigned)(ga >> 32) | (2u << 30);    // global_addr[56:32] | type=2
        asm volatile("tensor_load_to_lds %0, %1" :: "s"(g0), "s"(g1) : "memory");
    }

    for (int ks = 0; ks < kSteps; ++ks) {
        // ---- stage B chunk (im2col gather, zero-padded) ----
        {
            const int k0 = ks * 32 + kSub;
            v8bf bvals;
            #pragma unroll
            for (int i = 0; i < 8; ++i) {
                int k = k0 + i;
                float v = 0.0f;
                if (k < Ktrue) {
                    int ci = k / ktaps;
                    int dk = k - ci * ktaps;
                    int pos = tPos * stride + dk - pad;
                    if (pos >= 0 && pos < Lin)
                        v = (float)in[((long)nIdx * Cin + ci) * Lin + pos];
                }
                bvals[i] = (bf16)v;
            }
            *(v8bf*)&Bs[cLocal * 32 + kSub] = bvals;
        }
        // ---- TDM pipeline: issue next chunk, wait for current ----
        if (w == 0) {
            if (ks + 1 < kSteps) {
                unsigned long long ga =
                    (unsigned long long)(size_t)wts + (unsigned long long)(ks + 1) * 64u;
                u32x4 g0;
                g0[0] = 1u;
                g0[1] = ((ks + 1) & 1) ? ldsA1 : ldsA0;
                g0[2] = (unsigned)ga;
                g0[3] = (unsigned)(ga >> 32) | (2u << 30);
                asm volatile("tensor_load_to_lds %0, %1" :: "s"(g0), "s"(g1) : "memory");
                __builtin_amdgcn_s_wait_tensorcnt(1);  // oldest (chunk ks) done
            } else {
                __builtin_amdgcn_s_wait_tensorcnt(0);
            }
        }
        __syncthreads();

        const bf16* Acur = &As[ks & 1][0];

        // ---- B fragment: lane holds col (w&3)*16 + lane%16, K h*8..h*8+7 and 16+h*8..16+h*8+7 ----
        const bf16* pb = &Bs[((w & 3) * 16 + (lane & 15)) * 32];
        v8bf blo = *(const v8bf*)(pb + h * 8);
        v8bf bhi = *(const v8bf*)(pb + 16 + h * 8);
        v16bf bfrag;
        #pragma unroll
        for (int q = 0; q < 8; ++q) { bfrag[q] = blo[q]; bfrag[8 + q] = bhi[q]; }

        // ---- per M-tile: A fragment + WMMA (wave-uniform branches) ----
        #pragma unroll
        for (int j = 0; j < 4; ++j) {
            int mt = (w >> 2) + 2 * j;
            if (mt < mTiles) {
                const bf16* pa = &Acur[(mt * 16 + (lane & 15)) * 32];
                v8bf lo = *(const v8bf*)(pa + h * 8);        // K = h*8 .. h*8+7
                v8bf hi = *(const v8bf*)(pa + 16 + h * 8);   // K = 16+h*8 ..
                v16bf afrag;
                #pragma unroll
                for (int q = 0; q < 8; ++q) { afrag[q] = lo[q]; afrag[8 + q] = hi[q]; }
                acc[j] = __builtin_amdgcn_wmma_f32_16x16x32_bf16(
                    false, afrag, false, bfrag, (short)0, acc[j], false, false);
            }
        }
        __syncthreads();   // protects Bs rewrite and A buffer reuse next issue
    }

    // ---- epilogue: C/D layout -> lane holds col N=lane%16, rows r+8*h ----
    const int colL = (w & 3) * 16 + (lane & 15);
    const int gc   = colBase + colL;
    const int n2   = gc / Lout;
    const int t2   = gc - n2 * Lout;
    #pragma unroll
    for (int j = 0; j < 4; ++j) {
        int mt = (w >> 2) + 2 * j;
        if (mt < mTiles) {
            #pragma unroll
            for (int r = 0; r < 8; ++r) {
                int co = mt * 16 + r + 8 * h;
                long oidx = ((long)n2 * Cout + co) * Lout + t2;
                float v = acc[j][r];
                if (mode == 2) v += (float)ident[oidx];
                if (mode != 3) v = fmaxf(v, 0.0f);
                if (mode == 1 || mode == 2) {
                    float inv = rsqrtf(bnv[co] + 1e-5f);
                    v = (v - bnm[co]) * bng[co] * inv + bnb[co];
                }
                out[oidx] = (bf16)v;
            }
        }
    }
}

// ---------------------------------------------------------------------------
// Mean over time: act (nBeats,128,20) bf16 -> feat (nBeats,128) f32
// ---------------------------------------------------------------------------
__global__ void mean_time_kernel(const bf16* __restrict__ act, float* __restrict__ feat,
                                 int nBeats) {
    int i = blockIdx.x * 256 + threadIdx.x;
    if (i >= nBeats * 128) return;
    const bf16* p = act + (long)i * 20;
    float s = 0.f;
    #pragma unroll
    for (int t = 0; t < 20; ++t) s += (float)p[t];
    feat[i] = s * (1.0f / 20.0f);
}

// ---------------------------------------------------------------------------
// Beat attention pooling: feat viewed (nNodes, 9, 128); block=128 thr / signal
// ---------------------------------------------------------------------------
__global__ void beat_attn_kernel(const float* __restrict__ feat,
                                 const float* __restrict__ W,   // (128,64)
                                 const float* __restrict__ v,   // (64)
                                 float* __restrict__ pooled) {  // (nNodes,128)
    __shared__ float zb[TBEATS][64];
    __shared__ float sc[TBEATS];
    int sig = blockIdx.x, tid = threadIdx.x;
    const float* hrow = feat + (long)sig * TBEATS * 128;
    if (tid < 64) {
        for (int t = 0; t < TBEATS; ++t) {
            float z = 0.f;
            for (int c = 0; c < 128; ++c) z += hrow[t * 128 + c] * W[c * 64 + tid];
            zb[t][tid] = tanhf(z) * v[tid];
        }
    }
    __syncthreads();
    if (tid < TBEATS) {
        float s = 0.f;
        for (int j = 0; j < 64; ++j) s += zb[tid][j];
        sc[tid] = s;
    }
    __syncthreads();
    if (tid == 0) {
        float mx = -1e30f;
        for (int t = 0; t < TBEATS; ++t) mx = fmaxf(mx, sc[t]);
        float den = 0.f;
        for (int t = 0; t < TBEATS; ++t) { sc[t] = __expf(sc[t] - mx); den += sc[t]; }
        float inv = 1.0f / den;
        for (int t = 0; t < TBEATS; ++t) sc[t] *= inv;
    }
    __syncthreads();
    if (tid < 128) {
        float s = 0.f;
        for (int t = 0; t < TBEATS; ++t) s += hrow[t * 128 + tid] * sc[t];
        pooled[(long)sig * 128 + tid] = s;
    }
}

// ---------------------------------------------------------------------------
// GCN helpers
// ---------------------------------------------------------------------------
__global__ void fill_f32_kernel(float* p, float val, int n) {
    int i = blockIdx.x * 256 + threadIdx.x;
    if (i < n) p[i] = val;
}
__global__ void deg_kernel(const int* __restrict__ edge, float* deg, int E, int n) {
    int i = blockIdx.x * 256 + threadIdx.x;
    if (i >= E + n) return;
    int d = (i < E) ? edge[E + i] : (i - E);       // self loops appended
    atomicAdd(&deg[d], 1.0f);
}
__global__ void dis_kernel(const float* __restrict__ deg, float* dis, int n) {
    int i = blockIdx.x * 256 + threadIdx.x;
    if (i >= n) return;
    float d = deg[i];
    dis[i] = (d > 0.f) ? rsqrtf(d) : 0.f;
}
// C (n,64) = A (n,q) @ B (q,64); grid=n, block=64
__global__ void mm64_kernel(const float* __restrict__ A, const float* __restrict__ B,
                            float* __restrict__ C, int q) {
    int i = blockIdx.x, f = threadIdx.x;
    const float* a = A + (long)i * q;
    float s = 0.f;
    for (int c = 0; c < q; ++c) s += a[c] * B[c * 64 + f];
    C[(long)i * 64 + f] = s;
}
__global__ void init_bias_kernel(float* out, const float* __restrict__ b, int n) {
    int i = blockIdx.x * 256 + threadIdx.x;
    if (i < n * 64) out[i] = b[i & 63];
}
__global__ void gcn_agg_kernel(const float* __restrict__ h, const int* __restrict__ edge,
                               const float* __restrict__ dis, float* out, int E, int n) {
    long i = (long)blockIdx.x * 256 + threadIdx.x;
    long tot = (long)(E + n) * 64;
    if (i >= tot) return;
    int f = (int)(i & 63);
    int e = (int)(i >> 6);
    int s, d;
    if (e < E) { s = edge[e]; d = edge[E + e]; } else { s = d = e - E; }
    atomicAdd(&out[(long)d * 64 + f], h[(long)s * 64 + f] * dis[s] * dis[d]);
}
__global__ void relu_kernel(float* p, int n) {
    int i = blockIdx.x * 256 + threadIdx.x;
    if (i < n) p[i] = fmaxf(p[i], 0.f);
}

// ---------------------------------------------------------------------------
// Lead attention: xg (nGraphs,12,64), scores->softmax over leads, scale (no sum)
// ---------------------------------------------------------------------------
__global__ void lead_attn_kernel(const float* __restrict__ xg,
                                 const float* __restrict__ W,   // (64,64)
                                 const float* __restrict__ v,   // (64)
                                 float* __restrict__ xl2) {
    __shared__ float zb[LEADS][64];
    __shared__ float sc[LEADS];
    int g = blockIdx.x, tid = threadIdx.x;   // 64 threads
    const float* x = xg + (long)g * LEADS * 64;
    for (int l = 0; l < LEADS; ++l) {
        float z = 0.f;
        for (int c = 0; c < 64; ++c) z += x[l * 64 + c] * W[c * 64 + tid];
        zb[l][tid] = tanhf(z) * v[tid];
    }
    __syncthreads();
    if (tid < LEADS) {
        float s = 0.f;
        for (int j = 0; j < 64; ++j) s += zb[tid][j];
        sc[tid] = s;
    }
    __syncthreads();
    if (tid == 0) {
        float mx = -1e30f;
        for (int l = 0; l < LEADS; ++l) mx = fmaxf(mx, sc[l]);
        float den = 0.f;
        for (int l = 0; l < LEADS; ++l) { sc[l] = __expf(sc[l] - mx); den += sc[l]; }
        float inv = 1.0f / den;
        for (int l = 0; l < LEADS; ++l) sc[l] *= inv;
    }
    __syncthreads();
    for (int l = 0; l < LEADS; ++l)
        xl2[((long)g * LEADS + l) * 64 + tid] = x[l * 64 + tid] * sc[l];
}

// ---------------------------------------------------------------------------
// Head: per-graph max/mean pool -> relu(lin) -> fc; grid=nGraphs, block=128
// ---------------------------------------------------------------------------
__global__ void head_kernel(const float* __restrict__ xl2,
                            const float* __restrict__ linw, const float* __restrict__ linb,
                            const float* __restrict__ fcw,  const float* __restrict__ fcb,
                            float* __restrict__ out) {
    __shared__ float xc[128];
    __shared__ float hid[32];
    int g = blockIdx.x, tid = threadIdx.x;
    if (tid < 64) {
        float mx = -1e30f, s = 0.f;
        for (int l = 0; l < LEADS; ++l) {
            float v = xl2[((long)g * LEADS + l) * 64 + tid];
            mx = fmaxf(mx, v); s += v;
        }
        xc[tid] = mx;
        xc[64 + tid] = s * (1.0f / LEADS);
    }
    __syncthreads();
    if (tid < 32) {
        float s = linb[tid];
        for (int q = 0; q < 128; ++q) s += xc[q] * linw[q * 32 + tid];
        hid[tid] = fmaxf(s, 0.f);
    }
    __syncthreads();
    if (tid < 9) {
        float s = fcb[tid];
        for (int q = 0; q < 32; ++q) s += hid[q] * fcw[q * 9 + tid];
        out[(long)g * 9 + tid] = s;
    }
}

// ---------------------------------------------------------------------------
// Host orchestration
// ---------------------------------------------------------------------------
extern "C" void kernel_launch(void* const* d_in, const int* in_sizes, int n_in,
                              void* d_out, int out_size, void* d_ws, size_t ws_size,
                              hipStream_t stream) {
    const float* X    = (const float*)d_in[0];
    const int*   EDGE = (const int*)d_in[1];   // (2,E) int32; row0=src row1=dst
    // d_in[2] = batch (leads are contiguous groups of 12; not needed)

    const int E       = in_sizes[1] / 2;        // 16896
    const int nNodes  = in_sizes[0] / SIGLEN;   // 1536
    const int nGraphs = nNodes / LEADS;         // 128
    const int nBeats  = nNodes * TBEATS;        // 13824

    // params in setup_inputs() insertion order (recursively flattened)
    int pi = 3;
    auto P = [&]() -> const float* { return (const float*)d_in[pi++]; };
    const float* conv1w = P();
    const float *c1w[9], *c2w[9], *g1[9], *b1[9], *m1[9], *v1[9],
                *g2[9], *b2[9], *m2[9], *v2[9], *dwn[9];
    for (int i = 0; i < 9; ++i) {
        c1w[i] = P(); c2w[i] = P();
        g1[i] = P(); b1[i] = P(); m1[i] = P(); v1[i] = P();
        g2[i] = P(); b2[i] = P(); m2[i] = P(); v2[i] = P();
        dwn[i] = (i == 3 || i == 6) ? P() : nullptr;
    }
    const float* battW = P(); const float* battV = P();
    const float* gw1 = P(); const float* gb1 = P();
    const float* gw2 = P(); const float* gb2 = P();
    const float* gw3 = P(); const float* gb3 = P();
    const float* leadW = P(); const float* leadV = P();
    const float* linW = P(); const float* linB = P();
    const float* fcW = P(); const float* fcB = P();

    // workspace carve-out (~225 MB total)
    char* wp = (char*)d_ws;
    auto carve = [&](size_t bytes) -> void* {
        void* p = wp; wp += (bytes + 255) & ~(size_t)255; return p;
    };
    const size_t actElems = (size_t)nBeats * 32 * 80;  // constant across layers
    bf16*  seg    = (bf16*)carve((size_t)nBeats * BEATL * 2);
    bf16*  bufA   = (bf16*)carve(actElems * 2);
    bf16*  bufB   = (bf16*)carve(actElems * 2);
    bf16*  idB    = (bf16*)carve(actElems * 2);
    bf16*  wpk    = (bf16*)carve(2u * 1024 * 1024 * 2);  // packed weights pool
    float* feat   = (float*)carve((size_t)nBeats * 128 * 4);
    float* pooled = (float*)carve((size_t)nNodes * 128 * 4);
    float* hbuf   = (float*)carve((size_t)nNodes * 64 * 4);
    float* abuf   = (float*)carve((size_t)nNodes * 64 * 4);
    float* xl2    = (float*)carve((size_t)nNodes * 64 * 4);
    float* deg    = (float*)carve(nNodes * 4);
    float* dis    = (float*)carve(nNodes * 4);

    size_t woff = 0;
    auto pack = [&](const float* w, int Cout, int CinK) -> bf16* {
        int Kpad = (CinK + 31) & ~31;
        bf16* dst = wpk + woff;
        woff += (size_t)Cout * Kpad;
        int tot = Cout * Kpad;
        pack_w_kernel<<<(tot + 255) / 256, 256, 0, stream>>>(w, dst, Cout, CinK, Kpad);
        return dst;
    };
    auto conv = [&](const bf16* in, const bf16* w, const bf16* ident,
                    const float* bg, const float* bb, const float* bm, const float* bv,
                    bf16* out, int Cin, int Cout, int Li, int Lo,
                    int kt, int st, int pd, int Ktrue, int mode) {
        int Kpad = (Ktrue + 31) & ~31;
        int totalCols = nBeats * Lo;            // always a multiple of 64
        conv_wmma_kernel<<<totalCols / 64, 256, 0, stream>>>(
            in, w, ident, bg, bb, bm, bv, out,
            nBeats, Cin, Cout, Li, Lo, kt, st, pd, Ktrue, Kpad, mode);
    };

    // ---- CNN ----
    extract_beats_kernel<<<(nBeats * BEATL + 255) / 256, 256, 0, stream>>>(X, seg, nNodes);
    bf16* w0 = pack(conv1w, 32, 7);
    conv(seg, w0, nullptr, nullptr, nullptr, nullptr, nullptr,
         bufA, 1, 32, 80, 80, 7, 1, 3, 7, 0);

    static const int CI[9] = {32, 32, 32, 32, 64, 64, 64, 128, 128};
    static const int CO[9] = {32, 32, 32, 64, 64, 64, 128, 128, 128};
    static const int DS[9] = {0, 0, 0, 1, 0, 0, 1, 0, 0};
    bf16* cur = bufA;
    bf16* tmp = bufB;
    int Lin = 80;
    for (int i = 0; i < 9; ++i) {
        int k1 = DS[i] ? 8 : 7;
        int Lo = DS[i] ? Lin / 2 : Lin;
        bf16* w1p = pack(c1w[i], CO[i], CI[i] * k1);
        bf16* w2p = pack(c2w[i], CO[i], CO[i] * 7);
        const bf16* ident = cur;
        if (DS[i]) {
            bf16* wdp = pack(dwn[i], CO[i], CI[i] * 2);
            conv(cur, wdp, nullptr, nullptr, nullptr, nullptr, nullptr,
                 idB, CI[i], CO[i], Lin, Lo, 2, 2, 0, CI[i] * 2, 3);
            ident = idB;
        }
        conv(cur, w1p, nullptr, g1[i], b1[i], m1[i], v1[i],
             tmp, CI[i], CO[i], Lin, Lo, k1, DS[i] ? 2 : 1, 3, CI[i] * k1, 1);
        conv(tmp, w2p, ident, g2[i], b2[i], m2[i], v2[i],
             cur, CO[i], CO[i], Lo, Lo, 7, 1, 3, CO[i] * 7, 2);
        Lin = Lo;
    }

    // ---- time mean + beat attention ----
    mean_time_kernel<<<(nBeats * 128 + 255) / 256, 256, 0, stream>>>(cur, feat, nBeats);
    beat_attn_kernel<<<nNodes, 128, 0, stream>>>(feat, battW, battV, pooled);

    // ---- GCN normalization ----
    fill_f32_kernel<<<(nNodes + 255) / 256, 256, 0, stream>>>(deg, 0.f, nNodes);
    deg_kernel<<<(E + nNodes + 255) / 256, 256, 0, stream>>>(EDGE, deg, E, nNodes);
    dis_kernel<<<(nNodes + 255) / 256, 256, 0, stream>>>(deg, dis, nNodes);

    // ---- 3 GCN layers: h = x@W; agg = segsum(h[src]*norm)+b; relu ----
    long aggT = (long)(E + nNodes) * 64;
    int aggB = (int)((aggT + 255) / 256);
    mm64_kernel<<<nNodes, 64, 0, stream>>>(pooled, gw1, hbuf, 128);
    init_bias_kernel<<<(nNodes * 64 + 255) / 256, 256, 0, stream>>>(abuf, gb1, nNodes);
    gcn_agg_kernel<<<aggB, 256, 0, stream>>>(hbuf, EDGE, dis, abuf, E, nNodes);
    relu_kernel<<<(nNodes * 64 + 255) / 256, 256, 0, stream>>>(abuf, nNodes * 64);

    mm64_kernel<<<nNodes, 64, 0, stream>>>(abuf, gw2, hbuf, 64);
    init_bias_kernel<<<(nNodes * 64 + 255) / 256, 256, 0, stream>>>(abuf, gb2, nNodes);
    gcn_agg_kernel<<<aggB, 256, 0, stream>>>(hbuf, EDGE, dis, abuf, E, nNodes);
    relu_kernel<<<(nNodes * 64 + 255) / 256, 256, 0, stream>>>(abuf, nNodes * 64);

    mm64_kernel<<<nNodes, 64, 0, stream>>>(abuf, gw3, hbuf, 64);
    init_bias_kernel<<<(nNodes * 64 + 255) / 256, 256, 0, stream>>>(abuf, gb3, nNodes);
    gcn_agg_kernel<<<aggB, 256, 0, stream>>>(hbuf, EDGE, dis, abuf, E, nNodes);
    relu_kernel<<<(nNodes * 64 + 255) / 256, 256, 0, stream>>>(abuf, nNodes * 64);

    // ---- lead attention + head ----
    lead_attn_kernel<<<nGraphs, 64, 0, stream>>>(abuf, leadW, leadV, xl2);
    head_kernel<<<nGraphs, 128, 0, stream>>>(xl2, linW, linB, fcW, fcB, (float*)d_out);
}
